// Gated_DGL_58110907515590
// MI455X (gfx1250) — matrix operand order repaired and checked by reading, not compiled, not run on hardware
//
#include <hip/hip_runtime.h>
#include <hip/hip_bf16.h>

#define NN  100000
#define NE  1600000
#define HID 128

typedef __attribute__((ext_vector_type(16))) __bf16 v16bf;
typedef __attribute__((ext_vector_type(8)))  __bf16 v8bf;
typedef __attribute__((ext_vector_type(8)))  float  v8f;

union AB16 { v16bf v; v8bf h[2]; };

// ---- WMMA fragment loaders (CDNA5 16-bit layouts, wave32) ----
// A 16x32 bf16: lane l -> row l&15; K chunks [kg, kg+8) and [kg+16, kg+24), kg = (l&16)?8:0
__device__ __forceinline__ v16bf load_afrag(const __bf16* __restrict__ A, int rowBase, int k0, int lane) {
  const int m  = lane & 15;
  const int kg = (lane & 16) ? 8 : 0;
  const __bf16* p = A + (size_t)(rowBase + m) * HID + (k0 + kg);
  AB16 u;
  u.h[0] = *(const v8bf*)(p);
  u.h[1] = *(const v8bf*)(p + 16);
  return u.v;
}

// B 32x16 bf16 (B = W^T, W row-major [OUT][HID]): lane l -> col l&15,
// K chunks [ko, ko+8) and [ko+16, ko+24), ko = (l&16)?8:0  ->  W[col][k0+ko ..]
__device__ __forceinline__ v16bf load_bfrag(const __bf16* __restrict__ W, int colBase, int k0, int lane) {
  const int n  = lane & 15;
  const int ko = (lane & 16) ? 8 : 0;
  const __bf16* p = W + (size_t)(colBase + n) * HID + (k0 + ko);
  AB16 u;
  u.h[0] = *(const v8bf*)(p);
  u.h[1] = *(const v8bf*)(p + 16);
  return u.v;
}

__device__ __forceinline__ v8f wmma_bf16(v16bf a, v16bf b, v8f c) {
  return __builtin_amdgcn_wmma_f32_16x16x32_bf16(false, a, false, b, (short)0, c, false, false);
}

// ---- generic GEMM: C[N,outCols] = bf16(A[N,128]) @ bf16(W[outCols,128])^T + bias ----
// one wave per 16x16 tile; optional bf16 copy of output
__global__ void gemm_bias(const __bf16* __restrict__ A, const __bf16* __restrict__ W,
                          const float* __restrict__ bias, float* __restrict__ C,
                          __bf16* __restrict__ Cb, int outCols) {
  const int lane = threadIdx.x & 31;
  const int wave = threadIdx.x >> 5;
  const int rowBase = blockIdx.x * 16;
  const int colBase = wave * 16;
  if (colBase >= ((outCols + 15) & ~15)) return;   // wave-uniform exit (readout: 3 of 8 waves)

  v8f acc = {};
#pragma unroll
  for (int k0 = 0; k0 < HID; k0 += 32) {
    v16bf af = load_afrag(A, rowBase, k0, lane);
    v16bf bf = load_bfrag(W, colBase, k0, lane);
    acc = wmma_bf16(af, bf, acc);
  }
  const int col  = colBase + (lane & 15);
  const bool ok  = col < outCols;
  const float bv = ok ? bias[col] : 0.f;
  const int mB   = rowBase + ((lane & 16) ? 8 : 0);
#pragma unroll
  for (int v = 0; v < 8; ++v) {
    if (ok) {
      size_t idx = (size_t)(mB + v) * outCols + col;
      float val = acc[v] + bv;
      C[idx] = val;
      if (Cb) Cb[idx] = (__bf16)val;
    }
  }
}

// ---- fused GRU layer: gi = a@Wih^T+bih (384), gh = x@Whh^T+bhh (384), gates, ELU ----
// wave w of 8 handles feature cols [16w,16w+16); 6 accumulators cover r/z/n gate slices.
__global__ void gru_fused(const __bf16* __restrict__ Ab, const __bf16* __restrict__ Xb,
                          const __bf16* __restrict__ Wih, const __bf16* __restrict__ Whh,
                          const float* __restrict__ bih, const float* __restrict__ bhh,
                          const float* __restrict__ Xf, float* __restrict__ XfOut,
                          __bf16* __restrict__ XbOut) {
  const int lane = threadIdx.x & 31;
  const int wave = threadIdx.x >> 5;
  const int rowBase = blockIdx.x * 16;
  const int colBase = wave * 16;

  v8f air = {}, aiz = {}, ain = {}, ahr = {}, ahz = {}, ahn = {};
#pragma unroll
  for (int k0 = 0; k0 < HID; k0 += 32) {
    v16bf aA = load_afrag(Ab, rowBase, k0, lane);
    v16bf xA = load_afrag(Xb, rowBase, k0, lane);
    air = wmma_bf16(aA, load_bfrag(Wih, colBase,       k0, lane), air);
    aiz = wmma_bf16(aA, load_bfrag(Wih, colBase + 128, k0, lane), aiz);
    ain = wmma_bf16(aA, load_bfrag(Wih, colBase + 256, k0, lane), ain);
    ahr = wmma_bf16(xA, load_bfrag(Whh, colBase,       k0, lane), ahr);
    ahz = wmma_bf16(xA, load_bfrag(Whh, colBase + 128, k0, lane), ahz);
    ahn = wmma_bf16(xA, load_bfrag(Whh, colBase + 256, k0, lane), ahn);
  }
  const int col = colBase + (lane & 15);
  const float bir = bih[col], biz = bih[col + 128], bin = bih[col + 256];
  const float bhr = bhh[col], bhz = bhh[col + 128], bhn = bhh[col + 256];
  const int mB = rowBase + ((lane & 16) ? 8 : 0);
#pragma unroll
  for (int v = 0; v < 8; ++v) {
    size_t idx = (size_t)(mB + v) * HID + col;
    float hold = Xf[idx];
    float r = 1.f / (1.f + __expf(-(air[v] + bir + ahr[v] + bhr)));
    float z = 1.f / (1.f + __expf(-(aiz[v] + biz + ahz[v] + bhz)));
    float n = tanhf(ain[v] + bin + r * (ahn[v] + bhn));
    float xn = (1.f - z) * n + z * hold;
    float y = xn > 0.f ? xn : (__expf(xn) - 1.f);   // ELU(alpha=1)
    XfOut[idx] = y;
    XbOut[idx] = (__bf16)y;
  }
}

// ---- edge scatter: one wave per edge, lane l handles float4 chunk l ----
__global__ void scatter_edges(const float* __restrict__ m, const int* __restrict__ src,
                              const int* __restrict__ dst, float* __restrict__ a) {
  const int e    = (int)((blockIdx.x * blockDim.x + threadIdx.x) >> 5);
  const int lane = threadIdx.x & 31;
  if (e >= NE) return;
  const int s = src[e], d = dst[e];
  float4 val = ((const float4*)(m + (size_t)s * HID))[lane];
  float* ap = a + (size_t)d * HID + lane * 4;
  atomicAdd(ap + 0, val.x);
  atomicAdd(ap + 1, val.y);
  atomicAdd(ap + 2, val.z);
  atomicAdd(ap + 3, val.w);
}

__global__ void f32_to_bf16_pad(const float* __restrict__ in, __bf16* __restrict__ out,
                                int nsrc, int ntot) {
  int i = blockIdx.x * blockDim.x + threadIdx.x;
  if (i < ntot) out[i] = (i < nsrc) ? (__bf16)in[i] : (__bf16)0.f;
}

__global__ void zero_f32(float* __restrict__ p, int n) {
  int i = blockIdx.x * blockDim.x + threadIdx.x;
  if (i < n) p[i] = 0.f;
}

__global__ void log_softmax40(float* __restrict__ out) {
  int r = blockIdx.x * blockDim.x + threadIdx.x;
  if (r >= NN) return;
  float* p = out + (size_t)r * 40;
  float mx = -1e30f;
#pragma unroll
  for (int i = 0; i < 40; ++i) mx = fmaxf(mx, p[i]);
  float s = 0.f;
#pragma unroll
  for (int i = 0; i < 40; ++i) s += __expf(p[i] - mx);
  float lse = mx + __logf(s);
#pragma unroll
  for (int i = 0; i < 40; ++i) p[i] -= lse;
}

static inline int cdiv(long a, int b) { return (int)((a + b - 1) / b); }

extern "C" void kernel_launch(void* const* d_in, const int* in_sizes, int n_in,
                              void* d_out, int out_size, void* d_ws, size_t ws_size,
                              hipStream_t stream) {
  const float* h       = (const float*)d_in[0];
  const int*   src     = (const int*)d_in[1];
  const int*   dst     = (const int*)d_in[2];
  const float* embed_W = (const float*)d_in[4];
  const float* embed_b = (const float*)d_in[5];
  const float* read_W  = (const float*)d_in[6];
  const float* read_b  = (const float*)d_in[7];
  const float* eW[2]  = {(const float*)d_in[8],  (const float*)d_in[14]};
  const float* eb[2]  = {(const float*)d_in[9],  (const float*)d_in[15]};
  const float* Wih[2] = {(const float*)d_in[10], (const float*)d_in[16]};
  const float* bih[2] = {(const float*)d_in[11], (const float*)d_in[17]};
  const float* Whh[2] = {(const float*)d_in[12], (const float*)d_in[18]};
  const float* bhh[2] = {(const float*)d_in[13], (const float*)d_in[19]};

  // workspace layout
  char* w = (char*)d_ws;
  float*  x    = (float*)w;                 w += (size_t)NN * HID * 4;   // fp32 node state
  __bf16* xb0  = (__bf16*)w;                w += (size_t)NN * HID * 2;
  __bf16* xb1  = (__bf16*)w;                w += (size_t)NN * HID * 2;
  float*  m    = (float*)w;                 w += (size_t)NN * HID * 4;   // edge messages
  float*  a    = (float*)w;                 w += (size_t)NN * HID * 4;   // aggregated
  __bf16* ab   = (__bf16*)w;                w += (size_t)NN * HID * 2;   // bf16(a); also bf16(h)
  __bf16* wEmb = (__bf16*)w;                w += 128 * 128 * 2;
  __bf16* wE[2]  = {(__bf16*)w, (__bf16*)(w + 128 * 128 * 2)};          w += 2 * 128 * 128 * 2;
  __bf16* wIh[2] = {(__bf16*)w, (__bf16*)(w + 384 * 128 * 2)};          w += 2 * 384 * 128 * 2;
  __bf16* wHh[2] = {(__bf16*)w, (__bf16*)(w + 384 * 128 * 2)};          w += 2 * 384 * 128 * 2;
  __bf16* wRd  = (__bf16*)w;                w += 48 * 128 * 2;           // read_W zero-padded to 48 rows

  const int NH = NN * HID;
  const dim3 B(256);

  // converts (deterministic, every call)
  f32_to_bf16_pad<<<cdiv(NH, 256), B, 0, stream>>>(h, ab, NH, NH);
  f32_to_bf16_pad<<<cdiv(128 * 128, 256), B, 0, stream>>>(embed_W, wEmb, 128 * 128, 128 * 128);
  for (int l = 0; l < 2; ++l) {
    f32_to_bf16_pad<<<cdiv(128 * 128, 256), B, 0, stream>>>(eW[l],  wE[l],  128 * 128, 128 * 128);
    f32_to_bf16_pad<<<cdiv(384 * 128, 256), B, 0, stream>>>(Wih[l], wIh[l], 384 * 128, 384 * 128);
    f32_to_bf16_pad<<<cdiv(384 * 128, 256), B, 0, stream>>>(Whh[l], wHh[l], 384 * 128, 384 * 128);
  }
  f32_to_bf16_pad<<<cdiv(48 * 128, 256), B, 0, stream>>>(read_W, wRd, 40 * 128, 48 * 128);

  const int RT = NN / 16;  // 6250 row tiles

  // embed: x = h @ embed_W^T + b  (fp32 + bf16 copies)
  gemm_bias<<<RT, B, 0, stream>>>(ab, wEmb, embed_b, x, xb0, HID);

  __bf16* xin  = xb0;
  __bf16* xout = xb1;
  for (int l = 0; l < 2; ++l) {
    gemm_bias<<<RT, B, 0, stream>>>(xin, wE[l], eb[l], m, nullptr, HID);
    zero_f32<<<cdiv(NH, 256), B, 0, stream>>>(a, NH);
    scatter_edges<<<NE / 8, B, 0, stream>>>(m, src, dst, a);
    f32_to_bf16_pad<<<cdiv(NH, 256), B, 0, stream>>>(a, ab, NH, NH);
    gru_fused<<<RT, B, 0, stream>>>(ab, xin, wIh[l], wHh[l], bih[l], bhh[l], x, x, xout);
    __bf16* t = xin; xin = xout; xout = t;
  }

  // readout + log_softmax (in-place on d_out)
  gemm_bias<<<RT, B, 0, stream>>>(xin, wRd, read_b, (float*)d_out, nullptr, 40);
  log_softmax40<<<cdiv(NN, 256), B, 0, stream>>>((float*)d_out);

  (void)in_sizes; (void)n_in; (void)out_size; (void)ws_size;
}
